// Qwen3MoeSparseMoeBlock_20383914787231
// MI455X (gfx1250) — matrix operand order repaired and checked by reading, not compiled, not run on hardware
//
#include <hip/hip_runtime.h>
#include <hip/hip_bf16.h>

// ---------------- problem constants (from reference) ----------------
#define T_TOK 512
#define H_DIM 2048
#define I_DIM 768
#define N_EXP 32
#define TOPK  4
#define KC    256   // H-chunk staged in LDS per iteration (fp32)

typedef __attribute__((ext_vector_type(16))) __bf16 v16bf;
typedef __attribute__((ext_vector_type(8)))  __bf16 v8bf;
typedef __attribute__((ext_vector_type(8)))  float  v8f;

// --------------------------------------------------------------------
// CDNA5 async global->LDS copy (128-bit per lane), tracked on ASYNCcnt.
// VDST = per-lane LDS byte offset, VADDR = per-lane 64-bit global addr.
// Generic __shared__ pointers on amdgcn carry the LDS offset in the low
// 32 bits, so casting the pointer gives the VDST operand directly.
// --------------------------------------------------------------------
static __device__ __forceinline__ void async_load_lds_b128(void* lds_ptr,
                                                           const void* gptr) {
    uint32_t lds_off = (uint32_t)(uintptr_t)lds_ptr;
    uint64_t gaddr   = (uint64_t)(uintptr_t)gptr;
    asm volatile("global_load_async_to_lds_b128 %0, %1, off"
                 :: "v"(lds_off), "v"(gaddr)
                 : "memory");
}

static __device__ __forceinline__ void wait_asynccnt0() {
    asm volatile("s_wait_asynccnt 0x0" ::: "memory");
}

// =====================================================================
// Kernel 0: zero output accumulator + per-expert counters
// =====================================================================
__global__ void moe_zero(float* __restrict__ out, int* __restrict__ counts) {
    int i = blockIdx.x * 256 + threadIdx.x;
    if (i < T_TOK * H_DIM) out[i] = 0.0f;
    if (i < N_EXP) counts[i] = 0;
}

// =====================================================================
// Kernel 1: router. One wave32 per token, lane == expert (E == 32).
// softmax over lanes, iterative top-4 argmax, renormalize, scatter
// (token, weight) into per-expert lists.
// =====================================================================
__global__ __launch_bounds__(256)
void moe_router(const float* __restrict__ x, const float* __restrict__ gw,
                int* __restrict__ counts, int* __restrict__ tok_idx,
                float* __restrict__ tok_w) {
    const int wave = threadIdx.x >> 5;
    const int lane = threadIdx.x & 31;
    const int t = blockIdx.x * 8 + wave;
    if (t >= T_TOK) return;

    const float* __restrict__ xr = x  + (size_t)t    * H_DIM;
    const float* __restrict__ gr = gw + (size_t)lane * H_DIM;

    float acc = 0.0f;
#pragma unroll 4
    for (int h = 0; h < H_DIM; ++h)
        acc = fmaf(xr[h], gr[h], acc);

    // softmax over the 32 experts (one per lane)
    float m = acc;
#pragma unroll
    for (int off = 16; off; off >>= 1) m = fmaxf(m, __shfl_xor(m, off));
    float p = __expf(acc - m);
    float s = p;
#pragma unroll
    for (int off = 16; off; off >>= 1) s += __shfl_xor(s, off);
    p /= s;

    // top-4 by repeated argmax
    float wsel[TOPK];
    int   isel[TOPK];
    float tsum = 0.0f;
    float v = p;
#pragma unroll
    for (int k = 0; k < TOPK; ++k) {
        float mv = v;
        int   mi = lane;
#pragma unroll
        for (int off = 16; off; off >>= 1) {
            float ov = __shfl_xor(mv, off);
            int   oi = __shfl_xor(mi, off);
            if (ov > mv || (ov == mv && oi < mi)) { mv = ov; mi = oi; }
        }
        wsel[k] = mv;
        isel[k] = mi;
        tsum   += mv;
        if (lane == mi) v = -1.0f;   // remove winner
    }

    if (lane == 0) {
        float inv = 1.0f / tsum;     // norm_topk_prob
#pragma unroll
        for (int k = 0; k < TOPK; ++k) {
            int e    = isel[k];
            int slot = atomicAdd(&counts[e], 1);
            tok_idx[e * T_TOK + slot] = t;
            tok_w  [e * T_TOK + slot] = wsel[k] * inv;
        }
    }
}

// =====================================================================
// Kernel 2: per-expert gated MLP on gathered token tiles.
// grid = (token_tile, expert); block = 256 threads (8 waves).
// gate/up:  [16 x 2048] x [2048 x 768]  (48 n-tiles -> 6 per wave)
// down:     [16 x 768]  x [768 x 2048]  (128 n-tiles -> 16 per wave)
// x rows are gathered into LDS with GLOBAL_LOAD_ASYNC_TO_LDS_B128.
// All matmuls via v_wmma_f32_16x16x32_bf16, fp32 accumulate.
// =====================================================================
__global__ __launch_bounds__(256)
void moe_expert(const float* __restrict__ x,
                const float* __restrict__ wg,
                const float* __restrict__ wu,
                const float* __restrict__ wd,
                const int*   __restrict__ counts,
                const int*   __restrict__ tok_idx,
                const float* __restrict__ tok_w,
                float* __restrict__ out) {
    const int mt = blockIdx.x;            // token tile (tiles of same expert adjacent)
    const int e  = blockIdx.y;            // expert
    const int cnt = counts[e];
    if (mt * 16 >= cnt) return;

    __shared__ __align__(16) float  xsf[16][KC];     // gathered x chunk (fp32) 16 KB
    __shared__ __align__(16) __bf16 h2[16][I_DIM];   // silu(g)*u (bf16)        24 KB
    __shared__ int   s_tok[16];
    __shared__ float s_w[16];

    const int tid  = threadIdx.x;
    const int wave = tid >> 5;
    const int lane = tid & 31;
    const int row  = lane & 15;           // M row / N column within a 16-tile
    const int hi16 = lane >> 4;           // half-wave select (K split)

    if (tid < 16) {
        int slot = mt * 16 + tid;
        s_tok[tid] = (slot < cnt) ? tok_idx[e * T_TOK + slot] : tok_idx[e * T_TOK];
        s_w[tid]   = (slot < cnt) ? tok_w[e * T_TOK + slot] : 0.0f;
    }
    __syncthreads();

    const float* __restrict__ wgE = wg + (size_t)e * I_DIM * H_DIM;
    const float* __restrict__ wuE = wu + (size_t)e * I_DIM * H_DIM;
    const float* __restrict__ wdE = wd + (size_t)e * H_DIM * I_DIM;

    // -------------------- gate & up projections --------------------
    v8f accg[6], accu[6];
    {
        v8f z = {};
#pragma unroll
        for (int i = 0; i < 6; ++i) { accg[i] = z; accu[i] = z; }
    }

    for (int kc = 0; kc < H_DIM; kc += KC) {
        // async-gather x[16][KC] fp32 into LDS: 16*KC/4 = 1024 b128 moves,
        // 4 per thread, coalesced along each token row.
#pragma unroll
        for (int it = 0; it < 16 * (KC / 4) / 256; ++it) {
            int u  = it * 256 + tid;
            int r  = u / (KC / 4);
            int c  = (u % (KC / 4)) * 4;
            async_load_lds_b128(&xsf[r][c],
                                x + (size_t)s_tok[r] * H_DIM + kc + c);
        }
        wait_asynccnt0();     // my wave's async writes to LDS complete
        __syncthreads();      // everyone's complete

        for (int kk = 0; kk < KC; kk += 32) {
            // A fragment (16-bit A 16x32 layout): per lane two contiguous
            // 8-half chunks at K = off.. and K = off+16.. (off = hi16*8)
            const float* ap = &xsf[row][kk + hi16 * 8];
            v8f flo = *(const v8f*)(ap);
            v8f fhi = *(const v8f*)(ap + 16);
            v16bf af;
#pragma unroll
            for (int j = 0; j < 8; ++j) {
                af[j]     = (__bf16)flo[j];
                af[j + 8] = (__bf16)fhi[j];
            }

            const int kg = kc + kk;
#pragma unroll
            for (int nt = 0; nt < 6; ++nt) {
                const int n0 = (wave * 6 + nt) * 16;
                const float* bg = wgE + (size_t)(n0 + row) * H_DIM + kg + hi16 * 8;
                const float* bu = wuE + (size_t)(n0 + row) * H_DIM + kg + hi16 * 8;
                __builtin_prefetch(bg + 32, 0, 3);   // next K step, WGP-near
                __builtin_prefetch(bu + 32, 0, 3);
                v16bf bfg, bfu;
#pragma unroll
                for (int j = 0; j < 16; j += 4) {
                    const float4 vg = *(const float4*)(bg + j + (j & 8));
                    const float4 vu = *(const float4*)(bu + j + (j & 8));
                    bfg[j + 0] = (__bf16)vg.x; bfg[j + 1] = (__bf16)vg.y;
                    bfg[j + 2] = (__bf16)vg.z; bfg[j + 3] = (__bf16)vg.w;
                    bfu[j + 0] = (__bf16)vu.x; bfu[j + 1] = (__bf16)vu.y;
                    bfu[j + 2] = (__bf16)vu.z; bfu[j + 3] = (__bf16)vu.w;
                }
                accg[nt] = __builtin_amdgcn_wmma_f32_16x16x32_bf16(
                    false, af, false, bfg, (short)0, accg[nt], false, false);
                accu[nt] = __builtin_amdgcn_wmma_f32_16x16x32_bf16(
                    false, af, false, bfu, (short)0, accu[nt], false, false);
            }
        }
        __syncthreads();
    }

    // -------------------- fused SiLU(g) * u -> h2 (bf16, LDS) ------
#pragma unroll
    for (int nt = 0; nt < 6; ++nt) {
        const int ncol = (wave * 6 + nt) * 16 + row;
#pragma unroll
        for (int r = 0; r < 8; ++r) {
            const int m = r + hi16 * 8;
            const float g  = accg[nt][r];
            const float u  = accu[nt][r];
            const float sg = g / (1.0f + __expf(-g));
            h2[m][ncol] = (__bf16)(sg * u);
        }
    }
    __syncthreads();

    // -------------------- down projection --------------------------
    v8f accd[16];
    {
        v8f z = {};
#pragma unroll
        for (int i = 0; i < 16; ++i) accd[i] = z;
    }

    for (int kk = 0; kk < I_DIM; kk += 32) {
        const __bf16* ap = &h2[row][kk + hi16 * 8];
        v8bf alo = *(const v8bf*)(ap);
        v8bf ahi = *(const v8bf*)(ap + 16);
        v16bf af;
#pragma unroll
        for (int j = 0; j < 8; ++j) { af[j] = alo[j]; af[j + 8] = ahi[j]; }

#pragma unroll
        for (int nt = 0; nt < 16; ++nt) {
            const int n0 = (wave * 16 + nt) * 16;
            const float* bd = wdE + (size_t)(n0 + row) * I_DIM + kk + hi16 * 8;
            __builtin_prefetch(bd + 32, 0, 3);
            v16bf bfd;
#pragma unroll
            for (int j = 0; j < 16; j += 4) {
                const float4 vd = *(const float4*)(bd + j + (j & 8));
                bfd[j + 0] = (__bf16)vd.x; bfd[j + 1] = (__bf16)vd.y;
                bfd[j + 2] = (__bf16)vd.z; bfd[j + 3] = (__bf16)vd.w;
            }
            accd[nt] = __builtin_amdgcn_wmma_f32_16x16x32_bf16(
                false, af, false, bfd, (short)0, accd[nt], false, false);
        }
    }

    // -------------------- scale by routing weight, scatter-add -----
#pragma unroll
    for (int nt = 0; nt < 16; ++nt) {
        const int hcol = (wave * 16 + nt) * 16 + row;
#pragma unroll
        for (int r = 0; r < 8; ++r) {
            const int   m  = r + hi16 * 8;
            const float wv = s_w[m];
            if (wv != 0.0f) {
                unsafeAtomicAdd(&out[(size_t)s_tok[m] * H_DIM + hcol],
                                wv * accd[nt][r]);
            }
        }
    }
}

// =====================================================================
extern "C" void kernel_launch(void* const* d_in, const int* in_sizes, int n_in,
                              void* d_out, int out_size, void* d_ws, size_t ws_size,
                              hipStream_t stream) {
    const float* x      = (const float*)d_in[0];   // [T, H]
    const float* gw     = (const float*)d_in[1];   // [E, H]
    const float* wgate  = (const float*)d_in[2];   // [E, I, H]
    const float* wup    = (const float*)d_in[3];   // [E, I, H]
    const float* wdown  = (const float*)d_in[4];   // [E, H, I]
    float* out = (float*)d_out;

    // workspace layout
    int*   counts  = (int*)d_ws;                         // [E]
    int*   tok_idx = counts + N_EXP;                     // [E, T]
    float* tok_w   = (float*)(tok_idx + N_EXP * T_TOK);  // [E, T]

    moe_zero<<<(T_TOK * H_DIM + 255) / 256, 256, 0, stream>>>(out, counts);
    moe_router<<<T_TOK / 8, 256, 0, stream>>>(x, gw, counts, tok_idx, tok_w);

    dim3 grid(T_TOK / 16, N_EXP);   // x = token tile (same-expert tiles adjacent), y = expert
    moe_expert<<<grid, 256, 0, stream>>>(x, wgate, wup, wdown,
                                         counts, tok_idx, tok_w, out);
}
